// Model_25297357373596
// MI455X (gfx1250) — compile-verified, matrix operand order checked by reading, not run on hardware
//
// CDNA5 / gfx1250 implementation of the bidirectional-Mamba patch-TS model.
//
// Roofline: ~35 GFLOP total, dominated by the 336x5376x5376 head_gate GEMM and
// 14128-row x 128/256/512 encoder GEMMs. All weights (~130MB) fit in the 192MB
// L2; HBM traffic is negligible at 23.3 TB/s -> compute/latency bound. GEMMs
// use v_wmma_f32_16x16x32_f16 (f16 inputs staged through LDS, f32 accum); the
// sequential selective scan runs as one persistent block per batch element
// (channels in threads, 16 SSM states in registers, B/C via LDS broadcast).
//
// Staging invariant: every GEMM K and leading dimension in this model is a
// multiple of 4 and all base pointers are 16B-aligned, so float4 chunks along
// K are all-or-nothing valid -> single predicated global_load_b128 per chunk,
// ds_store_b64 of 4 converted halves (no scalar edge loop).
//
// Param flattening assumption (depth-first, dict insertion order):
//   d_in[0..3] = x, x_mark_enc, x_dec, x_mark_dec
//   4:view1.w 5:view1.b 6:view2.w 7:view2.b 8:view3.w 9:view3.b 10:view4.w
//   11:view4.b 12:W_P.w 13:W_P.b
//   14..65: inner_enc layers (26 ptrs/layer: mamba_f[9], mamba_r[9],
//           conv1.w/b, conv2.w/b, n1.g/b, n2.g/b), 66/67: inner norm g/b
//   68/69: head_gate w/b  70/71: head_lin w/b
//   72..123: outer_enc layers, 124/125 outer norm g/b, 126/127 nvh w/b
// Mamba block order: in_proj.w, conv_w, conv_b, x_proj.w, dt_proj.w,
//   dt_proj.b, A_log, D, out_proj.w

#include <hip/hip_runtime.h>

typedef __attribute__((ext_vector_type(16))) _Float16 v16h;
typedef __attribute__((ext_vector_type(4)))  _Float16 v4h;
typedef __attribute__((ext_vector_type(8)))  float    v8f;
typedef __attribute__((ext_vector_type(4)))  float    v4f;

#define B_      16
#define T_      336
#define N_      21
#define PLEN_   16
#define STR_    8
#define D_      128
#define PN_     42
#define NF_     5376
#define LIN_    883
#define LOUT_   22
#define PRED_   96

// ---------------------------------------------------------------------------
// Generic WMMA GEMM: C[M,N] = act(A[M,K] * W[N,K]^T + bias)
// 64x64 tile, 128 threads (4 waves), K-step 32, f16 staging in LDS, f32 accum.
// Requires K % 4 == 0, lda/ldw % 4 == 0, 16B-aligned A/W (true for all call
// sites in this model) so staging uses vector loads with a single predicate.
// ---------------------------------------------------------------------------
#define TM 64
#define TN 64
#define TK 32
#define LSTR 40   // padded LDS row stride in halves

__global__ __launch_bounds__(128) void k_gemm(
    const float* __restrict__ A, int lda,
    const float* __restrict__ W, int ldw,
    const float* __restrict__ bias,
    float* __restrict__ C, int ldc,
    int M, int Nn, int K, int act)
{
  __shared__ _Float16 sA[TM * LSTR];
  __shared__ _Float16 sB[TN * LSTR];
  const int tid  = threadIdx.x;
  const int wv   = tid >> 5;
  const int lane = tid & 31;
  const int hlf  = lane >> 4;     // which 16-lane half of the wave
  const int l16  = lane & 15;
  const int m_blk = blockIdx.y * TM;
  const int n_blk = blockIdx.x * TN;
  const int mrow0 = m_blk + wv * 16;

  v8f acc[4] = {};

  for (int k0 = 0; k0 < K; k0 += TK) {
    // stage A tile (64x32) as f16: 4 float4 chunks per thread, predicated
    #pragma unroll
    for (int i = tid * 4; i < TM * TK; i += 128 * 4) {
      int r = i >> 5, c = i & 31;          // c is a multiple of 4
      int gm = m_blk + r, gk = k0 + c;
      v4f v = {0.f, 0.f, 0.f, 0.f};
      if (gm < M && gk < K)                // K%4==0 -> whole chunk valid
        v = *reinterpret_cast<const v4f*>(&A[(size_t)gm * lda + gk]);
      v4h h;
      h[0] = (_Float16)v[0]; h[1] = (_Float16)v[1];
      h[2] = (_Float16)v[2]; h[3] = (_Float16)v[3];
      *reinterpret_cast<v4h*>(&sA[r * LSTR + c]) = h;
    }
    // stage W tile (64 output cols x 32 K) as f16
    #pragma unroll
    for (int i = tid * 4; i < TN * TK; i += 128 * 4) {
      int r = i >> 5, c = i & 31;
      int gn = n_blk + r, gk = k0 + c;
      v4f v = {0.f, 0.f, 0.f, 0.f};
      if (gn < Nn && gk < K)
        v = *reinterpret_cast<const v4f*>(&W[(size_t)gn * ldw + gk]);
      v4h h;
      h[0] = (_Float16)v[0]; h[1] = (_Float16)v[1];
      h[2] = (_Float16)v[2]; h[3] = (_Float16)v[3];
      *reinterpret_cast<v4h*>(&sB[r * LSTR + c]) = h;
    }
    // prefetch next K tile into cache while we compute this one
    if (k0 + TK < K) {
      int r = tid & 63;
      int gm = m_blk + r, gn = n_blk + r;
      if (gm < M)  __builtin_prefetch(&A[(size_t)gm * lda + k0 + TK], 0, 1);
      if (gn < Nn) __builtin_prefetch(&W[(size_t)gn * ldw + k0 + TK], 0, 1);
    }
    __syncthreads();

    // A fragment: lane half selects K sub-block; v16h element j -> K = j+(j&8)+8*hlf
    v16h af;
    {
      const _Float16* rowp = &sA[(wv * 16 + l16) * LSTR];
      #pragma unroll
      for (int j = 0; j < 16; ++j) af[j] = rowp[j + (j & 8) + (hlf << 3)];
    }
    #pragma unroll
    for (int t = 0; t < 4; ++t) {
      v16h bf;
      const _Float16* rowp = &sB[(t * 16 + l16) * LSTR];
      #pragma unroll
      for (int j = 0; j < 16; ++j) bf[j] = rowp[j + (j & 8) + (hlf << 3)];
      acc[t] = __builtin_amdgcn_wmma_f32_16x16x32_f16(
          false, af, false, bf, (short)0, acc[t], false, false);
    }
    __syncthreads();
  }

  // D layout: vgpr r, lane l -> M = r + 8*(l>=16), N = l&15
  #pragma unroll
  for (int t = 0; t < 4; ++t) {
    #pragma unroll
    for (int r = 0; r < 8; ++r) {
      int m = mrow0 + r + (hlf << 3);
      int n = n_blk + t * 16 + l16;
      if (m < M && n < Nn) {
        float v = acc[t][r];
        if (bias) v += bias[n];
        if (act == 1) v = v > 0.f ? v : 0.f;   // relu
        C[(size_t)m * ldc + n] = v;
      }
    }
  }
}

// ---------------------------------------------------------------------------
// Input normalization: per (b,n) mean/std over T, write z = (x-mean)/std
// ---------------------------------------------------------------------------
__global__ __launch_bounds__(128) void k_norm_input(
    const float* __restrict__ x, float* __restrict__ z,
    float* __restrict__ meanp, float* __restrict__ stdp)
{
  __shared__ float red[4];
  __shared__ float s_mean, s_inv;
  int bn = blockIdx.x;
  int b = bn / N_, n = bn % N_;
  float s = 0.f;
  for (int t = threadIdx.x; t < T_; t += 128)
    s += x[((size_t)b * T_ + t) * N_ + n];
  for (int o = 16; o > 0; o >>= 1) s += __shfl_down(s, o, 32);
  if ((threadIdx.x & 31) == 0) red[threadIdx.x >> 5] = s;
  __syncthreads();
  if (threadIdx.x == 0) s_mean = (red[0] + red[1] + red[2] + red[3]) / (float)T_;
  __syncthreads();
  float mean = s_mean;
  float s2 = 0.f;
  for (int t = threadIdx.x; t < T_; t += 128) {
    float d = x[((size_t)b * T_ + t) * N_ + n] - mean;
    s2 += d * d;
  }
  for (int o = 16; o > 0; o >>= 1) s2 += __shfl_down(s2, o, 32);
  if ((threadIdx.x & 31) == 0) red[threadIdx.x >> 5] = s2;
  __syncthreads();
  if (threadIdx.x == 0) {
    float var = (red[0] + red[1] + red[2] + red[3]) / (float)T_;
    float sd = sqrtf(var + 1e-5f);
    s_inv = 1.f / sd;
    meanp[bn] = mean; stdp[bn] = sd;
  }
  __syncthreads();
  float inv = s_inv;
  for (int t = threadIdx.x; t < T_; t += 128) {
    size_t idx = ((size_t)b * T_ + t) * N_ + n;
    z[idx] = (x[idx] - mean) * inv;
  }
}

// pv[b,t] = dot(z[b,t,:], w) + bias  (N=21 reduction)
__global__ void k_view_dot(const float* __restrict__ z, const float* __restrict__ w,
                           const float* __restrict__ bias, float* __restrict__ pv)
{
  int i = blockIdx.x * blockDim.x + threadIdx.x;
  if (i >= B_ * T_) return;
  const float* row = z + (size_t)i * N_;
  float s = bias[0];
  #pragma unroll
  for (int n = 0; n < N_; ++n) s += row[n] * w[n];
  pv[i] = s;
}

// patches[(b*N+n)*PN+p][i] = zc[b, n, p*8+i] (last col repeated for padding)
__global__ void k_patch(const float* __restrict__ z, float* __restrict__ patches, int total)
{
  int i = blockIdx.x * blockDim.x + threadIdx.x;
  if (i >= total) return;
  int pi = i & 15, rest = i >> 4;
  int p = rest % PN_; rest /= PN_;
  int n = rest % N_;  int b = rest / N_;
  int t = p * STR_ + pi; if (t >= T_) t = T_ - 1;
  patches[i] = z[((size_t)b * T_ + t) * N_ + n];
}

// u[b, 0..881, :] = embed[b*882 + r, :]; u[b, 882, :] = pview[b,:]
__global__ void k_assemble_u(const float* __restrict__ embed,
                             const float* __restrict__ pview,
                             float* __restrict__ u, int total)
{
  int i = blockIdx.x * blockDim.x + threadIdx.x;
  if (i >= total) return;
  int d = i & 127, row = i >> 7;
  int r = row % LIN_, b = row / LIN_;
  u[i] = (r < LIN_ - 1) ? embed[((size_t)b * (LIN_ - 1) + r) * D_ + d]
                        : pview[b * D_ + d];
}

// conv1d(D_CONV=2) + silu : xc[t] = silu(xr[t-1]*w0 + xr[t]*w1 + cb), xr = xz[:, :128]
__global__ void k_conv_silu(const float* __restrict__ xz, const float* __restrict__ cw,
                            const float* __restrict__ cb, float* __restrict__ xc,
                            int L, int total)
{
  int i = blockIdx.x * blockDim.x + threadIdx.x;
  if (i >= total) return;
  int d = i & 127, row = i >> 7, t = row % L;
  float cur  = xz[(size_t)row * 256 + d];
  float prev = (t > 0) ? xz[(size_t)(row - 1) * 256 + d] : 0.f;
  float v = prev * cw[d * 2 + 0] + cur * cw[d * 2 + 1] + cb[d];
  xc[i] = v / (1.f + __expf(-v));
}

__global__ void k_softplus(float* __restrict__ x, int total)
{
  int i = blockIdx.x * blockDim.x + threadIdx.x;
  if (i >= total) return;
  float v = x[i];
  x[i] = (v > 20.f) ? v : __logf(1.f + __expf(v));
}

// Selective scan: one block per batch, 128 threads = channels, 16 states/thread.
__global__ __launch_bounds__(128) void k_scan(
    const float* __restrict__ delta,  // (B*L,128)
    const float* __restrict__ dbc,    // (B*L,40): [8:24)=Bm, [24:40)=Cm
    const float* __restrict__ xc,     // (B*L,128)
    const float* __restrict__ A_log,  // (128,16)
    const float* __restrict__ Dp,     // (128)
    float* __restrict__ y, int L)
{
  __shared__ float sB[16], sC[16];
  const int b = blockIdx.x, d = threadIdx.x;
  float Aneg[16];
  #pragma unroll
  for (int s = 0; s < 16; ++s) Aneg[s] = -__expf(A_log[d * 16 + s]);
  const float Dd = Dp[d];
  float h[16] = {};
  for (int t = 0; t < L; ++t) {
    const size_t row = (size_t)b * L + t;
    if (d < 32) {
      float v = dbc[row * 40 + 8 + d];
      if (d < 16) sB[d] = v; else sC[d - 16] = v;
    }
    __syncthreads();
    const float dt = delta[row * 128 + d];
    const float xv = xc[row * 128 + d];
    float accv = 0.f;
    #pragma unroll
    for (int s = 0; s < 16; ++s) {
      float dA = __expf(dt * Aneg[s]);
      h[s] = dA * h[s] + dt * sB[s] * xv;
      accv += h[s] * sC[s];
    }
    y[row * 128 + d] = accv + xv * Dd;
    __syncthreads();
  }
}

// y *= silu(z) where z = xz[:, 128:256]
__global__ void k_gate(const float* __restrict__ xz, float* __restrict__ y, int total)
{
  int i = blockIdx.x * blockDim.x + threadIdx.x;
  if (i >= total) return;
  int d = i & 127, row = i >> 7;
  float zg = xz[(size_t)row * 256 + 128 + d];
  y[i] *= zg / (1.f + __expf(-zg));
}

// sequence reverse along L
__global__ void k_reverse(const float* __restrict__ in, float* __restrict__ out,
                          int L, int total)
{
  int i = blockIdx.x * blockDim.x + threadIdx.x;
  if (i >= total) return;
  int d = i & 127, row = i >> 7, t = row % L, b = row / L;
  out[i] = in[(((size_t)b * L + (L - 1 - t)) << 7) + d];
}

// LayerNorm(x [+ a] [+ b2]) over last dim 128; in-place safe.
__global__ __launch_bounds__(128) void k_ln(
    const float* __restrict__ x, const float* __restrict__ a,
    const float* __restrict__ b2,
    const float* __restrict__ g, const float* __restrict__ bt,
    float* __restrict__ out, int rows)
{
  __shared__ float red[4];
  __shared__ float s_mean, s_var;
  int row = blockIdx.x;
  if (row >= rows) return;
  int d = threadIdx.x;
  size_t idx = (size_t)row * 128 + d;
  float v = x[idx];
  if (a)  v += a[idx];
  if (b2) v += b2[idx];
  float s = v;
  for (int o = 16; o > 0; o >>= 1) s += __shfl_down(s, o, 32);
  if ((d & 31) == 0) red[d >> 5] = s;
  __syncthreads();
  if (d == 0) s_mean = (red[0] + red[1] + red[2] + red[3]) * (1.f / 128.f);
  __syncthreads();
  float mean = s_mean;
  float dv = (v - mean) * (v - mean);
  for (int o = 16; o > 0; o >>= 1) dv += __shfl_down(dv, o, 32);
  if ((d & 31) == 0) red[d >> 5] = dv;
  __syncthreads();
  if (d == 0) s_var = (red[0] + red[1] + red[2] + red[3]) * (1.f / 128.f);
  __syncthreads();
  out[idx] = (v - mean) * rsqrtf(s_var + 1e-5f) * g[d] + bt[d];
}

// flat[b,n, d*42+p] = x[(b*883 + n*42 + p)*128 + d]
__global__ void k_flatten(const float* __restrict__ x, float* __restrict__ flat, int total)
{
  int i = blockIdx.x * blockDim.x + threadIdx.x;
  if (i >= total) return;
  int c = i % NF_;
  int n = (i / NF_) % N_;
  int b = i / (NF_ * N_);
  int p = c % PN_, d = c / PN_;
  flat[i] = x[(((size_t)b * LIN_ + n * PN_ + p) << 7) + d];
}

// flat *= silu(gate)
__global__ void k_silu_mul(float* __restrict__ flat, const float* __restrict__ gate, int total)
{
  int i = blockIdx.x * blockDim.x + threadIdx.x;
  if (i >= total) return;
  float gv = gate[i];
  flat[i] *= gv / (1.f + __expf(-gv));
}

// outer input: (B,22,128) = concat(zh (B,21,128), nview (B,1,128))
__global__ void k_assemble_outer(const float* __restrict__ zh,
                                 const float* __restrict__ nview,
                                 float* __restrict__ xo, int total)
{
  int i = blockIdx.x * blockDim.x + threadIdx.x;
  if (i >= total) return;
  int d = i & 127, row = i >> 7;
  int n = row % LOUT_, b = row / LOUT_;
  xo[i] = (n < N_) ? zh[(((size_t)b * N_ + n) << 7) + d] : nview[(b << 7) + d];
}

// sumzo[b*21+n] = zh[b*21+n] + xouter[b*22+n]
__global__ void k_sum_zh_outer(const float* __restrict__ zh, const float* __restrict__ xo,
                               float* __restrict__ sumzo, int total)
{
  int i = blockIdx.x * blockDim.x + threadIdx.x;
  if (i >= total) return;
  int d = i & 127, row = i >> 7;
  int n = row % N_, b = row / N_;
  sumzo[i] = zh[i] + xo[(((size_t)b * LOUT_ + n) << 7) + d];
}

// out[b,t,n] = rr[(b*21+n)*96 + t] * std[b,n] + mean[b,n]
__global__ void k_final(const float* __restrict__ rr, const float* __restrict__ stdp,
                        const float* __restrict__ meanp, float* __restrict__ out, int total)
{
  int i = blockIdx.x * blockDim.x + threadIdx.x;
  if (i >= total) return;
  int n = i % N_;
  int t = (i / N_) % PRED_;
  int b = i / (N_ * PRED_);
  int bn = b * N_ + n;
  out[i] = rr[((size_t)bn) * PRED_ + t] * stdp[bn] + meanp[bn];
}

// ---------------------------------------------------------------------------
// Host-side orchestration
// ---------------------------------------------------------------------------
static inline void gemm(const float* A, int lda, const float* W, int ldw,
                        const float* bias, float* C, int ldc,
                        int M, int Nn, int K, int act, hipStream_t s)
{
  dim3 g((Nn + TN - 1) / TN, (M + TM - 1) / TM);
  k_gemm<<<g, 128, 0, s>>>(A, lda, W, ldw, bias, C, ldc, M, Nn, K, act);
}

#define EWL(n) <<<((n) + 255) / 256, 256, 0, stream>>>

struct Scr {
  float* xrev; float* mf; float* mr;
  float* xz;   float* xc; float* dbc; float* delta; float* yscan; float* ffn;
};

static void run_mamba(const float* xin, int Bsz, int L, const float* const* mp,
                      Scr& S, float* outp, hipStream_t stream)
{
  const int BL = Bsz * L, tot = BL * D_;
  gemm(xin, D_, mp[0], D_, nullptr, S.xz, 2 * D_, BL, 2 * D_, D_, 0, stream);  // in_proj
  k_conv_silu EWL(tot)(S.xz, mp[1], mp[2], S.xc, L, tot);
  gemm(S.xc, D_, mp[3], D_, nullptr, S.dbc, 40, BL, 40, D_, 0, stream);        // x_proj
  gemm(S.dbc, 40, mp[4], 8, mp[5], S.delta, D_, BL, D_, 8, 0, stream);         // dt_proj
  k_softplus EWL(tot)(S.delta, tot);
  k_scan<<<Bsz, 128, 0, stream>>>(S.delta, S.dbc, S.xc, mp[6], mp[7], S.yscan, L);
  k_gate EWL(tot)(S.xz, S.yscan, tot);
  gemm(S.yscan, D_, mp[8], D_, nullptr, outp, D_, BL, D_, D_, 0, stream);      // out_proj
}

static void run_encoder(float* x, int Bsz, int L, const float* const* IN, int base,
                        Scr& S, hipStream_t stream)
{
  const int BL = Bsz * L, tot = BL * D_;
  for (int l = 0; l < 2; ++l) {
    const float* const* lp = IN + base + l * 26;
    run_mamba(x, Bsz, L, lp + 0, S, S.mf, stream);          // forward mamba
    k_reverse EWL(tot)(x, S.xrev, L, tot);
    run_mamba(S.xrev, Bsz, L, lp + 9, S, S.xrev, stream);   // reverse mamba
    k_reverse EWL(tot)(S.xrev, S.mr, L, tot);
    k_ln<<<BL, 128, 0, stream>>>(x, S.mf, S.mr, lp[22], lp[23], x, BL);  // norm1
    gemm(x, D_, lp[18], D_, lp[19], S.ffn, 512, BL, 512, D_, 1, stream); // conv1+relu
    gemm(S.ffn, 512, lp[20], 512, lp[21], S.yscan, D_, BL, D_, 512, 0, stream); // conv2
    k_ln<<<BL, 128, 0, stream>>>(x, S.yscan, nullptr, lp[24], lp[25], x, BL);   // norm2
  }
  k_ln<<<BL, 128, 0, stream>>>(x, nullptr, nullptr, IN[base + 52], IN[base + 53], x, BL);
}

extern "C" void kernel_launch(void* const* d_in, const int* in_sizes, int n_in,
                              void* d_out, int out_size, void* d_ws, size_t ws_size,
                              hipStream_t stream)
{
  (void)in_sizes; (void)out_size; (void)ws_size;
  const float* IN[160];
  for (int i = 0; i < n_in && i < 160; ++i) IN[i] = (const float*)d_in[i];
  float* out = (float*)d_out;

  float* ws = (float*)d_ws;
  size_t off = 0;
  auto alloc = [&](size_t n) {
    float* p = ws + off;
    off += (n + 255) & ~(size_t)255;
    return p;
  };

  const int BLI = B_ * LIN_;                       // 14128 token rows
  float* z       = alloc((size_t)B_ * T_ * N_);
  float* meanp   = alloc(B_ * N_);
  float* stdp    = alloc(B_ * N_);
  float* pv      = alloc(B_ * T_);
  float* nv      = alloc(B_ * T_);
  float* pview   = alloc(B_ * D_);
  float* nview   = alloc(B_ * D_);
  float* patches = alloc((size_t)B_ * N_ * PN_ * PLEN_);
  float* embed   = alloc((size_t)B_ * N_ * PN_ * D_);   // reused as `flat`
  float* x       = alloc((size_t)BLI * D_);
  float* xrev    = alloc((size_t)BLI * D_);
  float* mf      = alloc((size_t)BLI * D_);
  float* mr      = alloc((size_t)BLI * D_);
  float* xc      = alloc((size_t)BLI * D_);
  float* delta   = alloc((size_t)BLI * D_);
  float* yscan   = alloc((size_t)BLI * D_);
  float* big     = alloc((size_t)BLI * 512);            // xz / ffn / gate (phase-shared)
  float* dbc     = alloc((size_t)BLI * 40);
  float* zh      = alloc((size_t)B_ * N_ * D_);
  float* sumzo   = alloc((size_t)B_ * N_ * D_);
  float* rr      = alloc((size_t)B_ * N_ * PRED_);

  // 1) instance norm over T
  k_norm_input<<<B_ * N_, 128, 0, stream>>>(IN[0], z, meanp, stdp);

  // 2) pview / nview
  k_view_dot EWL(B_ * T_)(z, IN[4], IN[5], pv);
  k_view_dot EWL(B_ * T_)(z, IN[8], IN[9], nv);
  gemm(pv, T_, IN[6],  T_, IN[7],  pview, D_, B_, D_, T_, 0, stream);
  gemm(nv, T_, IN[10], T_, IN[11], nview, D_, B_, D_, T_, 0, stream);

  // 3) patching + patch embedding
  {
    int tot = B_ * N_ * PN_ * PLEN_;
    k_patch EWL(tot)(z, patches, tot);
  }
  gemm(patches, PLEN_, IN[12], PLEN_, IN[13], embed, D_, B_ * N_ * PN_, D_, PLEN_, 0, stream);
  k_assemble_u EWL(BLI * D_)(embed, pview, x, BLI * D_);

  // 4) inner encoder (L=883)
  Scr S{xrev, mf, mr, big, xc, dbc, delta, yscan, big};
  run_encoder(x, B_, LIN_, IN, 14, S, stream);

  // 5) head: flatten -> gate (336x5376x5376 WMMA GEMM) -> head_lin
  {
    int tot = B_ * N_ * NF_;
    k_flatten EWL(tot)(x, embed, tot);                         // flat
    float* gate = big;
    gemm(embed, NF_, IN[68], NF_, IN[69], gate, NF_, B_ * N_, NF_, NF_, 0, stream);
    k_silu_mul EWL(tot)(embed, gate, tot);
    gemm(embed, NF_, IN[70], NF_, IN[71], zh, D_, B_ * N_, D_, NF_, 0, stream);
  }

  // 6) outer encoder (L=22)
  k_assemble_outer EWL(B_ * LOUT_ * D_)(zh, nview, x, B_ * LOUT_ * D_);
  run_encoder(x, B_, LOUT_, IN, 72, S, stream);

  // 7) final head + denormalize
  k_sum_zh_outer EWL(B_ * N_ * D_)(zh, x, sumzo, B_ * N_ * D_);
  gemm(sumzo, D_, IN[126], D_, IN[127], rr, PRED_, B_ * N_, PRED_, D_, 0, stream);
  k_final EWL(B_ * PRED_ * N_)(rr, stdp, meanp, out, B_ * PRED_ * N_);
}
